// NoiseRobustGINClassifier_64536178590373
// MI455X (gfx1250) — matrix operand order repaired and checked by reading, not compiled, not run on hardware
//
#include <hip/hip_runtime.h>
#include <hip/hip_bf16.h>

typedef __attribute__((ext_vector_type(16))) __bf16 v16bf;
typedef __attribute__((ext_vector_type(8)))  float  v8f;

#define GIN_N 100000
#define GIN_E 1600000
#define GIN_G 512
#define GIN_H 128
#define GIN_L 5

#define GBM 128
#define GBN 64
#define GBK 32
#define LDA (GBK + 4)   // padded LDS row stride (bf16 units)
#define LDB (GBK + 4)

// ---- fragment loaders (layouts per CDNA5 ISA 7.12.2) -----------------------
__device__ inline v16bf ldfragA(const __bf16* p) {
  // lane half holds K = {kb..kb+7} in e0..e7 and {kb+16..kb+23} in e8..e15
  v16bf r;
#pragma unroll
  for (int j = 0; j < 8; ++j) { r[j] = p[j]; r[j + 8] = p[j + 16]; }
  return r;
}
__device__ inline v16bf ldfragB(const __bf16* p) {
  // lane half holds K = {kb..kb+7} in e0..e7 and {kb+16..kb+23} in e8..e15
  v16bf r;
#pragma unroll
  for (int j = 0; j < 8; ++j) { r[j] = p[j]; r[j + 8] = p[j + 16]; }
  return r;
}

union Pack4 { __bf16 b[4]; uint2 u; };

// ---- fused GEMM: out = epi(A[M,K] @ W[K,Nc]) with bf16x3 WMMA --------------
// epilogue: y = (acc + bias - mean)*gamma*rsqrt(var+eps) + beta ; relu ; +res
__global__ __launch_bounds__(256)
void gemm_bn_relu_kernel(const float* __restrict__ A, const float* __restrict__ W,
                         const float* __restrict__ bias, const float* __restrict__ bnp,
                         const float* __restrict__ res, float* __restrict__ out,
                         int M, int K, int Nc, int relu) {
  __shared__ __bf16 Ah[GBM * LDA];
  __shared__ __bf16 Al[GBM * LDA];
  __shared__ __bf16 Bh[GBN * LDB];
  __shared__ __bf16 Bl[GBN * LDB];

  const int tid  = threadIdx.x;
  const int bm   = blockIdx.y * GBM;
  const int bn0  = blockIdx.x * GBN;
  const int wave = tid >> 5;
  const int lane = tid & 31;
  const int wm = (wave >> 1) * 32;   // 4 waves tiling M
  const int wn = (wave & 1) * 32;    // 2 waves tiling N
  const int lrow = lane & 15;
  const int lhi  = lane >> 4;        // 0: lanes 0-15, 1: lanes 16-31
  const int kbA = lhi * 8;           // A-fragment K base
  const int kbB = lhi * 8;           // B-fragment K base
  const bool fullM = (bm + GBM) <= M;

  // staging thread mapping
  const int a_c4 = (tid & 7) * 4;    // A: k offset {0,4,...,28}
  const int a_m0 = tid >> 3;         // A: row 0..31 (4 passes of 32 rows)
  const int b_n4 = (tid & 15) * 4;   // B: n offset {0,4,...,60}
  const int b_k0 = tid >> 4;         // B: k 0..15 (2 passes)

  v8f acc[4];
#pragma unroll
  for (int t = 0; t < 4; ++t)
#pragma unroll
    for (int j = 0; j < 8; ++j) acc[t][j] = 0.f;

  for (int k0 = 0; k0 < K; k0 += GBK) {
    __syncthreads();
    // stage A tile (GBM x GBK) as hi/lo bf16 planes, float4 loads
    if (fullM) {
#pragma unroll
      for (int i = 0; i < 4; ++i) {
        int m = a_m0 + i * 32;
        const float4 v = *(const float4*)&A[(size_t)(bm + m) * K + k0 + a_c4];
        Pack4 ph, pl;
        const float f[4] = {v.x, v.y, v.z, v.w};
#pragma unroll
        for (int j = 0; j < 4; ++j) {
          ph.b[j] = (__bf16)f[j];
          pl.b[j] = (__bf16)(f[j] - (float)ph.b[j]);
        }
        *(uint2*)&Ah[m * LDA + a_c4] = ph.u;
        *(uint2*)&Al[m * LDA + a_c4] = pl.u;
      }
    } else {
#pragma unroll
      for (int i = 0; i < 4; ++i) {
        int m = a_m0 + i * 32;
        int gm = bm + m;
        float4 v = make_float4(0.f, 0.f, 0.f, 0.f);
        if (gm < M) v = *(const float4*)&A[(size_t)gm * K + k0 + a_c4];
        Pack4 ph, pl;
        const float f[4] = {v.x, v.y, v.z, v.w};
#pragma unroll
        for (int j = 0; j < 4; ++j) {
          ph.b[j] = (__bf16)f[j];
          pl.b[j] = (__bf16)(f[j] - (float)ph.b[j]);
        }
        *(uint2*)&Ah[m * LDA + a_c4] = ph.u;
        *(uint2*)&Al[m * LDA + a_c4] = pl.u;
      }
    }
    // stage B tile transposed: Bh[n*LDB + k] = W[k0+k][bn0+n], float4 loads
#pragma unroll
    for (int i = 0; i < 2; ++i) {
      int kk = b_k0 + i * 16;
      const float4 v = *(const float4*)&W[(size_t)(k0 + kk) * Nc + bn0 + b_n4];
      const float f[4] = {v.x, v.y, v.z, v.w};
#pragma unroll
      for (int j = 0; j < 4; ++j) {
        __bf16 hi = (__bf16)f[j];
        Bh[(b_n4 + j) * LDB + kk] = hi;
        Bl[(b_n4 + j) * LDB + kk] = (__bf16)(f[j] - (float)hi);
      }
    }
    // prefetch next K-tile while this one computes (global_prefetch_b8)
    if (k0 + GBK < K) {
      __builtin_prefetch(&A[(size_t)(bm + a_m0) * K + k0 + GBK + a_c4], 0, 1);
      __builtin_prefetch(&W[(size_t)(k0 + GBK + b_k0) * Nc + bn0 + b_n4], 0, 1);
    }
    __syncthreads();

    v16bf afh[2], afl[2], bfh[2], bfl[2];
#pragma unroll
    for (int mi = 0; mi < 2; ++mi) {
      int r = wm + mi * 16 + lrow;
      afh[mi] = ldfragA(&Ah[r * LDA + kbA]);
      afl[mi] = ldfragA(&Al[r * LDA + kbA]);
    }
#pragma unroll
    for (int ni = 0; ni < 2; ++ni) {
      int c = wn + ni * 16 + lrow;
      bfh[ni] = ldfragB(&Bh[c * LDB + kbB]);
      bfl[ni] = ldfragB(&Bl[c * LDB + kbB]);
    }
#pragma unroll
    for (int mi = 0; mi < 2; ++mi)
#pragma unroll
      for (int ni = 0; ni < 2; ++ni) {
        v8f c = acc[mi * 2 + ni];
        c = __builtin_amdgcn_wmma_f32_16x16x32_bf16(false, afh[mi], false, bfh[ni], (short)0, c, false, false);
        c = __builtin_amdgcn_wmma_f32_16x16x32_bf16(false, afh[mi], false, bfl[ni], (short)0, c, false, false);
        c = __builtin_amdgcn_wmma_f32_16x16x32_bf16(false, afl[mi], false, bfh[ni], (short)0, c, false, false);
        acc[mi * 2 + ni] = c;
      }
  }

  // epilogue
#pragma unroll
  for (int mi = 0; mi < 2; ++mi)
#pragma unroll
    for (int ni = 0; ni < 2; ++ni) {
      int n = bn0 + wn + ni * 16 + lrow;
      float bb = bias ? bias[n] : 0.f;
      float ls = 1.f, lo = bb;
      if (bnp) {
        float ga = bnp[n], be = bnp[Nc + n], mu = bnp[2 * Nc + n], va = bnp[3 * Nc + n];
        ls = ga * rsqrtf(va + 1e-5f);
        lo = (bb - mu) * ls + be;
      }
      int rbase = bm + wm + mi * 16 + lhi * 8;
      v8f c = acc[mi * 2 + ni];
#pragma unroll
      for (int j = 0; j < 8; ++j) {
        int r = rbase + j;
        if (r < M) {
          float y = c[j] * ls + lo;
          if (relu) y = fmaxf(y, 0.f);
          if (res) y += res[(size_t)r * Nc + n];
          out[(size_t)r * Nc + n] = y;
        }
      }
    }
}

// ---- agg init: agg = (1 + eps[layer]) * h ----------------------------------
__global__ void scale_init_kernel(const float* __restrict__ h, const float* __restrict__ eps,
                                  int layer, float* __restrict__ agg, int n) {
  int i = blockIdx.x * blockDim.x + threadIdx.x;
  if (i < n) agg[i] = (1.f + eps[layer]) * h[i];
}

// ---- edge scatter: agg[dst] += h[src] (4 feats / thread) -------------------
__global__ void scatter_kernel(const float* __restrict__ h, const int* __restrict__ ei,
                               float* __restrict__ agg) {
  int idx = blockIdx.x * blockDim.x + threadIdx.x;
  if (idx >= GIN_E * (GIN_H / 4)) return;
  int e = idx >> 5;               // H/4 == 32 chunks
  int c = (idx & 31) * 4;
  int src = ei[e];
  int dst = ei[GIN_E + e];
  const float4 v = *(const float4*)&h[(size_t)src * GIN_H + c];
  float* d = &agg[(size_t)dst * GIN_H + c];
  atomicAdd(d + 0, v.x);
  atomicAdd(d + 1, v.y);
  atomicAdd(d + 2, v.z);
  atomicAdd(d + 3, v.w);
}

// ---- pooling ---------------------------------------------------------------
__global__ void zero_pool_kernel(float* __restrict__ s, float* __restrict__ mx,
                                 int* __restrict__ cnt) {
  int i = blockIdx.x * blockDim.x + threadIdx.x;
  if (i < GIN_G * GIN_H) { s[i] = 0.f; mx[i] = 0.f; }
  if (i < GIN_G) cnt[i] = 0;
}

__global__ void pool_kernel(const float* __restrict__ h, const int* __restrict__ batch,
                            float* __restrict__ s, unsigned* __restrict__ mx,
                            int* __restrict__ cnt) {
  int idx = blockIdx.x * blockDim.x + threadIdx.x;
  if (idx >= GIN_N * GIN_H) return;
  int n = idx >> 7;               // H == 128
  int f = idx & 127;
  int g = batch[n];
  float v = h[idx];
  atomicAdd(&s[(size_t)g * GIN_H + f], v);
  // h >= 0 (ReLU chains + non-negative residuals) => bit pattern is monotone
  atomicMax(&mx[(size_t)g * GIN_H + f], __float_as_uint(v));
  if (f == 0) atomicAdd(&cnt[g], 1);
}

__global__ void build_g_kernel(const float* __restrict__ s, const float* __restrict__ mx,
                               const int* __restrict__ cnt, float* __restrict__ g) {
  int idx = blockIdx.x * blockDim.x + threadIdx.x;
  if (idx >= GIN_G * GIN_H) return;
  int gi = idx >> 7;
  int f = idx & 127;
  float sv = s[idx];
  float c = (float)cnt[gi];
  float mean = sv / fmaxf(c, 1.f);
  float mv = mx[idx];  // init 0 == where(cnt>0, mx, 0) for non-negative h
  float* row = &g[(size_t)gi * 3 * GIN_H];
  row[f] = sv;
  row[GIN_H + f] = mean;
  row[2 * GIN_H + f] = mv;
}

// ---- tiny classifier head: [G,64] @ [64,6] + b -----------------------------
__global__ void head_kernel(const float* __restrict__ f2, const float* __restrict__ W,
                            const float* __restrict__ b, float* __restrict__ out) {
  int idx = blockIdx.x * blockDim.x + threadIdx.x;
  if (idx >= GIN_G * 6) return;
  int g = idx / 6;
  int o = idx - g * 6;
  float acc = b[o];
#pragma unroll
  for (int k = 0; k < 64; ++k) acc += f2[g * 64 + k] * W[k * 6 + o];
  out[idx] = acc;
}

// ---------------------------------------------------------------------------
static inline void launch_gemm(const float* A, const float* W, const float* bias,
                               const float* bnp, const float* res, float* out,
                               int M, int K, int Nc, int relu, hipStream_t stream) {
  dim3 grid((Nc + GBN - 1) / GBN, (M + GBM - 1) / GBM);
  gemm_bn_relu_kernel<<<grid, 256, 0, stream>>>(A, W, bias, bnp, res, out, M, K, Nc, relu);
}

extern "C" void kernel_launch(void* const* d_in, const int* in_sizes, int n_in,
                              void* d_out, int out_size, void* d_ws, size_t ws_size,
                              hipStream_t stream) {
  (void)in_sizes; (void)n_in; (void)out_size; (void)ws_size;
  const int H = GIN_H;
  const float* x        = (const float*)d_in[0];
  const int*   ei       = (const int*)d_in[1];
  const int*   batch    = (const int*)d_in[2];
  const float* in_W     = (const float*)d_in[3];
  const float* in_b     = (const float*)d_in[4];
  const float* in_bn    = (const float*)d_in[5];
  const float* W1       = (const float*)d_in[6];
  const float* b1       = (const float*)d_in[7];
  const float* bn1      = (const float*)d_in[8];
  const float* W2       = (const float*)d_in[9];
  const float* b2       = (const float*)d_in[10];
  const float* bn2      = (const float*)d_in[11];
  const float* W3       = (const float*)d_in[12];
  const float* b3       = (const float*)d_in[13];
  const float* eps_g    = (const float*)d_in[14];
  const float* node_bn  = (const float*)d_in[15];
  const float* cls_W1   = (const float*)d_in[16];
  const float* cls_b1   = (const float*)d_in[17];
  const float* cls_bn1  = (const float*)d_in[18];
  const float* cls_W2   = (const float*)d_in[19];
  const float* cls_b2   = (const float*)d_in[20];
  const float* cls_bn2  = (const float*)d_in[21];
  const float* fin_W    = (const float*)d_in[22];
  const float* fin_b    = (const float*)d_in[23];

  float* ws = (float*)d_ws;
  size_t NH = (size_t)GIN_N * H;
  size_t GH = (size_t)GIN_G * H;
  float* h    = ws;                     // [N,H]
  float* agg  = ws + NH;                // [N,H]
  float* t3   = ws + 2 * NH;            // [N,3H]
  float* t2   = ws + 5 * NH;            // [N,2H]
  float* s    = ws + 7 * NH;            // [G,H]
  float* mx   = s + GH;                 // [G,H]
  int*   cnt  = (int*)(mx + GH);        // [G]
  float* gbuf = (float*)(cnt + GIN_G);  // [G,3H]
  float* f1   = gbuf + 3 * GH;          // [G,H]
  float* f2   = f1 + GH;                // [G,64]

  // input projection: h = relu(bn(x @ in_W + in_b))
  launch_gemm(x, in_W, in_b, in_bn, nullptr, h, GIN_N, 128, H, 1, stream);

  const int nh = (int)NH;
  for (int i = 0; i < GIN_L; ++i) {
    scale_init_kernel<<<(nh + 255) / 256, 256, 0, stream>>>(h, eps_g, i, agg, nh);
    scatter_kernel<<<(GIN_E * (GIN_H / 4) + 255) / 256, 256, 0, stream>>>(h, ei, agg);
    launch_gemm(agg, W1 + (size_t)i * H * 3 * H, b1 + (size_t)i * 3 * H,
                bn1 + (size_t)i * 4 * 3 * H, nullptr, t3, GIN_N, H, 3 * H, 1, stream);
    launch_gemm(t3, W2 + (size_t)i * 3 * H * 2 * H, b2 + (size_t)i * 2 * H,
                bn2 + (size_t)i * 4 * 2 * H, nullptr, t2, GIN_N, 3 * H, 2 * H, 1, stream);
    launch_gemm(t2, W3 + (size_t)i * 2 * H * H, b3 + (size_t)i * H,
                node_bn + (size_t)i * 4 * H, (i ? h : nullptr), h, GIN_N, 2 * H, H, 1, stream);
  }

  zero_pool_kernel<<<((int)(GH) + 255) / 256, 256, 0, stream>>>(s, mx, cnt);
  pool_kernel<<<(GIN_N * GIN_H + 255) / 256, 256, 0, stream>>>(h, batch, s, (unsigned*)mx, cnt);
  build_g_kernel<<<((int)(GH) + 255) / 256, 256, 0, stream>>>(s, mx, cnt, gbuf);

  launch_gemm(gbuf, cls_W1, cls_b1, cls_bn1, nullptr, f1, GIN_G, 3 * H, H, 1, stream);
  launch_gemm(f1, cls_W2, cls_b2, cls_bn2, nullptr, f2, GIN_G, H, 64, 1, stream);
  head_kernel<<<(GIN_G * 6 + 255) / 256, 256, 0, stream>>>(f2, fin_W, fin_b, (float*)d_out);
}
